// TriplaneNeRF_31636729102599
// MI455X (gfx1250) — matrix operand order repaired and checked by reading, not compiled, not run on hardware
//
#include <hip/hip_runtime.h>

// ---------------------------------------------------------------------------
// TriplaneNeRF fused kernel for gfx1250 (CDNA5, wave32, WMMA 16x16x32 f16)
// - f16 weights transposed + staged once per block in (dynamic) LDS
// - all 5 linear layers (incl. 1-wide / 3-wide heads) run on the WMMA pipe
// - 88 v_wmma per 16-point tile; LDS-only operand path
// ---------------------------------------------------------------------------

typedef __attribute__((ext_vector_type(16))) _Float16 v16h;
typedef __attribute__((ext_vector_type(8)))  _Float16 v8h;
typedef __attribute__((ext_vector_type(8)))  float    v8f;

#define RR    64      // plane resolution
#define CC    32      // channels per plane
#define HH    128     // hidden width
#define FEATN 96      // 3*CC
#define FPAD  104     // padded feat row (halves): 208B rows, 16B aligned
#define WPAD  136     // padded weight/h row (halves): 272B rows, 16B aligned
#define HDPAD 24      // padded head-weight row (halves): 48B rows, 16B aligned
#define WAVES 8
#define TPB   (WAVES * 32)

// ---- WMMA wrapper: D = A(16x32 f16) * B(32x16 f16) + C(16x16 f32) ----------
static __device__ inline v8f wmma_f16(v16h a, v16h b, v8f c) {
  return __builtin_amdgcn_wmma_f32_16x16x32_f16(
      /*neg_a=*/false, a, /*neg_b=*/false, b,
      /*c_mod=*/(short)0, c, /*reuse_a=*/false, /*reuse_b=*/false);
}

// ---- A-fragment loader (16-bit A 16x32 ISA layout) -------------------------
// lane L: M = L%16 ; kbase = (L/16)*8 ; halves = K[kbase..+7], K[kbase+16..+23]
static __device__ inline v16h load_a_frag(const _Float16* row, int kc, int kbase) {
  v8h lo = *(const v8h*)(row + kc * 32 + kbase);
  v8h hi = *(const v8h*)(row + kc * 32 + kbase + 16);
  return __builtin_shufflevector(lo, hi,
      0, 1, 2, 3, 4, 5, 6, 7, 8, 9, 10, 11, 12, 13, 14, 15);
}

// ---- B-fragment loader from LDS-staged transposed f16 weight Wt[k][n] ------
// Dense 16-bit B (32x16, KxN): lane L holds column N=L%16,
// halves = K[(L/16)*8..+7], K[(L/16)*8+16..+23] (strided gather).
static __device__ inline v16h load_b_lds(const _Float16* wt, int ldn, int ntile,
                                         int kc, int lane) {
  const _Float16* p = wt + (size_t)(kc * 32 + (lane >> 4) * 8) * ldn + ntile * 16 + (lane & 15);
  v16h b;
#pragma unroll
  for (int s = 0; s < 8; ++s) {
    b[s] = p[s * ldn];
    b[8 + s] = p[(16 + s) * ldn];
  }
  return b;
}

__global__ __launch_bounds__(TPB) void triplane_nerf_kernel(
    const float* __restrict__ points, const float* __restrict__ planes,
    const float* __restrict__ dW1, const float* __restrict__ db1,
    const float* __restrict__ dW2, const float* __restrict__ db2,
    const float* __restrict__ dW3, const float* __restrict__ db3,
    const float* __restrict__ rW1, const float* __restrict__ rb1,
    const float* __restrict__ rW2, const float* __restrict__ rb2,
    float* __restrict__ out_rgb, float* __restrict__ out_den, int total_pts) {
  // ---- static LDS: per-wave tiles + bias vectors (~61.5 KB) ---------------
  __shared__ _Float16 featLds[WAVES][16][FPAD];  // 26624 B
  __shared__ _Float16 hLds[WAVES][16][WPAD];     // 34816 B
  __shared__ float s_db1[HH], s_db2[HH], s_rb1[HH];
  __shared__ float s_db3, s_rb2v[3];

  // ---- dynamic LDS: transposed f16 weights, rows padded -------------------
  extern __shared__ _Float16 dynw[];
  _Float16* wt1  = dynw;                   // [96][WPAD]   : dW1^T
  _Float16* wt2  = wt1 + FEATN * WPAD;     // [128][WPAD]  : dW2^T
  _Float16* wtr1 = wt2 + HH * WPAD;        // [96][WPAD]   : rW1^T
  _Float16* whd  = wtr1 + FEATN * WPAD;    // [128][HDPAD] : [dW3 | rW2^T | 0]

  // ---- one-time staging: f32 -> f16, transpose to [in][out] ---------------
  for (int i = threadIdx.x; i < FEATN * HH; i += TPB) {
    const int k = i >> 7, n = i & (HH - 1);
    wt1[k * WPAD + n]  = (_Float16)dW1[n * FEATN + k];
    wtr1[k * WPAD + n] = (_Float16)rW1[n * FEATN + k];
  }
  for (int i = threadIdx.x; i < HH * HH; i += TPB) {
    const int k = i >> 7, n = i & (HH - 1);
    wt2[k * WPAD + n] = (_Float16)dW2[n * HH + k];
  }
  // head tile: col 0 = dW3, cols 1..3 = rW2 rows, cols 4..15 = 0
  for (int i = threadIdx.x; i < HH * 16; i += TPB) {
    const int k = i >> 4, n = i & 15;
    float v = 0.0f;
    if (n == 0)      v = dW3[k];
    else if (n < 4)  v = rW2[(n - 1) * HH + k];
    whd[k * HDPAD + n] = (_Float16)v;
  }
  for (int i = threadIdx.x; i < HH; i += TPB) {
    s_db1[i] = db1[i];
    s_db2[i] = db2[i];
    s_rb1[i] = rb1[i];
  }
  if (threadIdx.x == 0) {
    s_db3 = db3[0];
    s_rb2v[0] = rb2[0]; s_rb2v[1] = rb2[1]; s_rb2v[2] = rb2[2];
  }
  __syncthreads();

  const int lane  = threadIdx.x & 31;
  const int wave  = threadIdx.x >> 5;
  const int ptl   = lane & 15;     // point-in-tile / D-frag column
  const int khalf = lane >> 4;
  const int kbase = khalf * 8;     // A-fragment K base
  const int rbase = khalf * 8;     // D-fragment M base

  const int nTiles = total_pts >> 4;          // 16 points per wave-tile
  const int tileStride = gridDim.x * WAVES;

  for (int tile = blockIdx.x * WAVES + wave; tile < nTiles; tile += tileStride) {
    // prefetch next tile's points into the cache hierarchy
    {
      const int nxt = tile + tileStride;
      if (nxt < nTiles)
        __builtin_prefetch(&points[3 * (nxt * 16 + ptl)], 0, 3);
    }

    // ======================================================================
    // 1. Tri-plane bilinear gather -> featLds (f16), lane = (point, ch-half)
    // ======================================================================
    {
      const int g = tile * 16 + ptl;
      const float px = points[3 * g + 0];
      const float py = points[3 * g + 1];
      const float pz = points[3 * g + 2];
      const float cx[3] = {px, px, py};
      const float cy[3] = {py, pz, pz};
#pragma unroll
      for (int pl = 0; pl < 3; ++pl) {
        float ix = (cx[pl] + 1.0f) * 0.5f * (float)(RR - 1);
        float iy = (cy[pl] + 1.0f) * 0.5f * (float)(RR - 1);
        ix = fminf(fmaxf(ix, 0.0f), (float)(RR - 1));
        iy = fminf(fmaxf(iy, 0.0f), (float)(RR - 1));
        const int x0 = (int)floorf(ix);
        const int y0 = (int)floorf(iy);
        const int x1 = (x0 + 1 < RR - 1) ? x0 + 1 : RR - 1;
        const int y1 = (y0 + 1 < RR - 1) ? y0 + 1 : RR - 1;
        const float wx = ix - (float)x0, wy = iy - (float)y0;
        const float w00 = (1.f - wx) * (1.f - wy), w01 = wx * (1.f - wy);
        const float w10 = (1.f - wx) * wy,         w11 = wx * wy;
#pragma unroll
        for (int c = 0; c < 16; ++c) {
          const int ch = khalf * 16 + c;
          const float* p0 = planes + ((size_t)(pl * CC + ch) * RR + y0) * RR;
          const float* p1 = planes + ((size_t)(pl * CC + ch) * RR + y1) * RR;
          const float v = p0[x0] * w00 + p0[x1] * w01 +
                          p1[x0] * w10 + p1[x1] * w11;
          featLds[wave][ptl][pl * CC + ch] = (_Float16)v;
        }
      }
    }
    asm volatile("s_wait_dscnt 0" ::: "memory");  // drain writes before A reads

    // ======================================================================
    // 2. A-fragments of feat (16x96), reused by density & rgb branches
    // ======================================================================
    const _Float16* frow = &featLds[wave][ptl][0];
    const v16h fa0 = load_a_frag(frow, 0, kbase);
    const v16h fa1 = load_a_frag(frow, 1, kbase);
    const v16h fa2 = load_a_frag(frow, 2, kbase);

    // ======================================================================
    // 3. Density layer 1: h = relu(feat @ dW1^T + db1)   (24 WMMAs)
    // ======================================================================
    for (int nt = 0; nt < HH / 16; ++nt) {
      v8f acc = {0.f, 0.f, 0.f, 0.f, 0.f, 0.f, 0.f, 0.f};
      acc = wmma_f16(fa0, load_b_lds(wt1, WPAD, nt, 0, lane), acc);
      acc = wmma_f16(fa1, load_b_lds(wt1, WPAD, nt, 1, lane), acc);
      acc = wmma_f16(fa2, load_b_lds(wt1, WPAD, nt, 2, lane), acc);
      const int col = nt * 16 + ptl;
      const float bias = s_db1[col];
#pragma unroll
      for (int r = 0; r < 8; ++r) {
        const float v = fmaxf(acc[r] + bias, 0.0f);
        hLds[wave][rbase + r][col] = (_Float16)v;
      }
    }
    asm volatile("s_wait_dscnt 0" ::: "memory");

    // ======================================================================
    // 4. Density layer 2: h2 = relu(h @ dW2^T + db2)     (32 WMMAs)
    // ======================================================================
    const _Float16* hrow = &hLds[wave][ptl][0];
    {
      const v16h ha0 = load_a_frag(hrow, 0, kbase);
      const v16h ha1 = load_a_frag(hrow, 1, kbase);
      const v16h ha2 = load_a_frag(hrow, 2, kbase);
      const v16h ha3 = load_a_frag(hrow, 3, kbase);
      for (int nt = 0; nt < HH / 16; ++nt) {
        v8f acc = {0.f, 0.f, 0.f, 0.f, 0.f, 0.f, 0.f, 0.f};
        acc = wmma_f16(ha0, load_b_lds(wt2, WPAD, nt, 0, lane), acc);
        acc = wmma_f16(ha1, load_b_lds(wt2, WPAD, nt, 1, lane), acc);
        acc = wmma_f16(ha2, load_b_lds(wt2, WPAD, nt, 2, lane), acc);
        acc = wmma_f16(ha3, load_b_lds(wt2, WPAD, nt, 3, lane), acc);
        const int col = nt * 16 + ptl;
        const float bias = s_db2[col];
#pragma unroll
        for (int r = 0; r < 8; ++r) {
          const float v = fmaxf(acc[r] + bias, 0.0f);
          hLds[wave][rbase + r][col] = (_Float16)v;  // overwrite h with h2
        }
      }
    }
    asm volatile("s_wait_dscnt 0" ::: "memory");

    // ======================================================================
    // 5. Density head on WMMA: col 0 of [dW3|rW2|0] tile    (4 WMMAs)
    // ======================================================================
    {
      const v16h ga0 = load_a_frag(hrow, 0, kbase);
      const v16h ga1 = load_a_frag(hrow, 1, kbase);
      const v16h ga2 = load_a_frag(hrow, 2, kbase);
      const v16h ga3 = load_a_frag(hrow, 3, kbase);
      v8f acc = {0.f, 0.f, 0.f, 0.f, 0.f, 0.f, 0.f, 0.f};
      acc = wmma_f16(ga0, load_b_lds(whd, HDPAD, 0, 0, lane), acc);
      acc = wmma_f16(ga1, load_b_lds(whd, HDPAD, 0, 1, lane), acc);
      acc = wmma_f16(ga2, load_b_lds(whd, HDPAD, 0, 2, lane), acc);
      acc = wmma_f16(ga3, load_b_lds(whd, HDPAD, 0, 3, lane), acc);
      if (ptl == 0) {  // lanes 0 and 16 hold the density column
        const float b3 = s_db3;
#pragma unroll
        for (int r = 0; r < 8; ++r) {
          const float x = acc[r] + b3;
          const float d = (x > 20.0f) ? x : log1pf(__expf(x));
          out_den[tile * 16 + rbase + r] = d;
        }
      }
    }

    // ======================================================================
    // 6. RGB layer 1: r = relu(feat @ rW1^T + rb1)       (24 WMMAs)
    // ======================================================================
    for (int nt = 0; nt < HH / 16; ++nt) {
      v8f acc = {0.f, 0.f, 0.f, 0.f, 0.f, 0.f, 0.f, 0.f};
      acc = wmma_f16(fa0, load_b_lds(wtr1, WPAD, nt, 0, lane), acc);
      acc = wmma_f16(fa1, load_b_lds(wtr1, WPAD, nt, 1, lane), acc);
      acc = wmma_f16(fa2, load_b_lds(wtr1, WPAD, nt, 2, lane), acc);
      const int col = nt * 16 + ptl;
      const float bias = s_rb1[col];
#pragma unroll
      for (int r = 0; r < 8; ++r) {
        const float v = fmaxf(acc[r] + bias, 0.0f);
        hLds[wave][rbase + r][col] = (_Float16)v;  // overwrite h2 with r
      }
    }
    asm volatile("s_wait_dscnt 0" ::: "memory");

    // ======================================================================
    // 7. RGB head on WMMA: cols 1..3 of [dW3|rW2|0] tile     (4 WMMAs)
    // ======================================================================
    {
      const v16h ra0 = load_a_frag(hrow, 0, kbase);
      const v16h ra1 = load_a_frag(hrow, 1, kbase);
      const v16h ra2 = load_a_frag(hrow, 2, kbase);
      const v16h ra3 = load_a_frag(hrow, 3, kbase);
      v8f acc = {0.f, 0.f, 0.f, 0.f, 0.f, 0.f, 0.f, 0.f};
      acc = wmma_f16(ra0, load_b_lds(whd, HDPAD, 0, 0, lane), acc);
      acc = wmma_f16(ra1, load_b_lds(whd, HDPAD, 0, 1, lane), acc);
      acc = wmma_f16(ra2, load_b_lds(whd, HDPAD, 0, 2, lane), acc);
      acc = wmma_f16(ra3, load_b_lds(whd, HDPAD, 0, 3, lane), acc);
      if (ptl >= 1 && ptl <= 3) {  // lanes 1..3 / 17..19 hold rgb columns
        const int j = ptl - 1;
        const float bj = s_rb2v[j];
#pragma unroll
        for (int r = 0; r < 8; ++r) {
          const float x = acc[r] + bj;
          out_rgb[3 * (tile * 16 + rbase + r) + j] = 1.0f / (1.0f + __expf(-x));
        }
      }
    }
  }
}

extern "C" void kernel_launch(void* const* d_in, const int* in_sizes, int n_in,
                              void* d_out, int out_size, void* d_ws,
                              size_t ws_size, hipStream_t stream) {
  (void)n_in; (void)out_size; (void)d_ws; (void)ws_size;
  const float* points = (const float*)d_in[0];
  const float* planes = (const float*)d_in[1];
  const float* dW1 = (const float*)d_in[2];
  const float* db1 = (const float*)d_in[3];
  const float* dW2 = (const float*)d_in[4];
  const float* db2 = (const float*)d_in[5];
  const float* dW3 = (const float*)d_in[6];
  const float* db3 = (const float*)d_in[7];
  const float* rW1 = (const float*)d_in[8];
  const float* rb1 = (const float*)d_in[9];
  const float* rW2 = (const float*)d_in[10];
  const float* rb2 = (const float*)d_in[11];

  const int total_pts = in_sizes[0] / 3;  // B*N
  float* out_rgb = (float*)d_out;
  float* out_den = out_rgb + (size_t)3 * total_pts;

  const int nTiles = total_pts >> 4;
  int blocks = (nTiles + WAVES - 1) / WAVES;
  if (blocks > 1024) blocks = 1024;

  // dynamic LDS: 3 transposed weight matrices + head tile (f16)
  const size_t dynBytes =
      ((size_t)(FEATN + HH + FEATN) * WPAD + (size_t)HH * HDPAD) *
      sizeof(_Float16);

  triplane_nerf_kernel<<<blocks, TPB, dynBytes, stream>>>(
      points, planes, dW1, db1, dW2, db2, dW3, db3, rW1, rb1, rW2, rb2,
      out_rgb, out_den, total_pts);
}
